// Model_50379966382300
// MI455X (gfx1250) — matrix operand order repaired and checked by reading, not compiled, not run on hardware
//
#include <hip/hip_runtime.h>
#include <hip/hip_bf16.h>
#include <cfloat>

// ---------------------------------------------------------------------------
// Informer forward (ProbSparse transformer) for gfx1250 / MI455X.
//  * All dense GEMMs run on v_wmma_f32_16x16x32_f16 (f16 in, f32 accumulate).
//  * GEMM: one block = 128(M) x 128(N) tile; B slab double-buffered in LDS,
//    staged with GLOBAL_LOAD_ASYNC_TO_LDS_B128 (ASYNCcnt + s_wait_asynccnt),
//    shared by 8 waves; B-fragment ds_loads pipelined against WMMA. One
//    barrier per K-step.
//  * GEMM epilogue can emit an f16 copy (C16) so downstream GEMMs need no
//    separate conversion pass; f32 output is optional (FFN1 emits f16 only).
//  * ProbSparse machinery (M-measure, top-k, softmax, cumsum/mean ctx,
//    scatter) is VALU/LDS work with wave32 shuffle reductions.
//  * Input pointer indices assume jax tree_util sorted-key flattening of the
//    setup_inputs() dict: params leaves first (keys sorted recursively), then
//    x_dec, x_enc, x_mark_dec, x_mark_enc.
// ---------------------------------------------------------------------------

static constexpr int CB   = 4;     // batch
static constexpr int CL   = 2048;  // sequence length (enc & dec)
static constexpr int CD   = 256;   // d_model
static constexpr int CH   = 8;     // heads
static constexpr int CDK  = 32;    // head dim
static constexpr int CDF  = 1024;  // d_ff
static constexpr int CTOK = CB * CL;   // 8192 rows
static constexpr int CBH  = CB * CH;   // 32 batch*head
static constexpr int CUU  = 40;    // active queries (factor*ceil(ln L))
static constexpr int CUP  = 48;    // padded to 16-multiple for WMMA
static constexpr int CPRED = 1024;
static constexpr int CCO  = 7;

typedef __attribute__((ext_vector_type(16))) _Float16 v16h;
typedef __attribute__((ext_vector_type(8)))  _Float16 v8h;
typedef __attribute__((ext_vector_type(8)))  float    v8f;

__device__ __forceinline__ v8f wmma_f16(v16h a, v16h b, v8f c) {
  return __builtin_amdgcn_wmma_f32_16x16x32_f16(false, a, false, b, (short)0, c,
                                                false, false);
}

__device__ __forceinline__ v16h cat16(v8h lo, v8h hi) {
  return __builtin_shufflevector(lo, hi, 0,1,2,3,4,5,6,7,8,9,10,11,12,13,14,15);
}

__device__ __forceinline__ float gelu_exact(float x) {
  return 0.5f * x * (1.0f + erff(x * 0.70710678118654752f));
}

__device__ __forceinline__ unsigned mix_hash(unsigned x) {
  x ^= x >> 16; x *= 0x7feb352dU;
  x ^= x >> 15; x *= 0x846ca68bU;
  x ^= x >> 16; return x;
}

// Async copy of 32 bytes (per lane) from global to LDS. INST_OFFSET is added
// to both the LDS destination and the global source (ISA 08_async_tensor §4.4),
// so the same operands serve both halves. Tracked by ASYNCcnt.
__device__ __forceinline__ void async_copy32(unsigned lds_byte,
                                             const void* gsrc) {
  unsigned long long g = (unsigned long long)(size_t)gsrc;
  asm volatile(
      "global_load_async_to_lds_b128 %0, %1, off\n\t"
      "global_load_async_to_lds_b128 %0, %1, off offset:16"
      :: "v"(lds_byte), "v"(g) : "memory");
}

__device__ __forceinline__ void wait_async0() {
  asm volatile("s_wait_asynccnt 0x0" ::: "memory");
}

// ---------------------------------------------------------------------------
// Generic GEMM: out = act( scale*(A16[M][K] @ BT16[N][K]^T) + bias + resid )
// Block = 128x128 tile (8 waves x [16x128] strips); B slab double-buffered in
// LDS via async loads. Requires M%128==0, N%128==0, K%32==0 (true here).
// C (f32) and C16 (f16) outputs are each optional (may be null).
// ---------------------------------------------------------------------------
__global__ __launch_bounds__(256) void gemm_wmma_kernel(
    const _Float16* __restrict__ A, int lda,
    const _Float16* __restrict__ BT, int ldb,
    const float* __restrict__ bias,
    const float* __restrict__ resid,
    float* __restrict__ C, _Float16* __restrict__ C16, int ldc,
    int M, int N, int K, float scale, int do_gelu)
{
  constexpr int SLAB = 128 * 32;            // halfs per slab (8 KB)
  __shared__ _Float16 bs[2 * SLAB];         // 16 KB double buffer

  const int tid   = threadIdx.x;
  const int wavei = tid >> 5;               // 0..7
  const int lane  = tid & 31;
  const int nstrips = N >> 7;
  const int mb = blockIdx.x / nstrips;
  const int st = blockIdx.x - mb * nstrips;
  const int l15 = lane & 15;
  const int kh  = lane >> 4;
  const int mt  = (mb << 3) + wavei;
  const size_t arow = (size_t)(mt * 16 + l15) * (size_t)lda;

  // Cooperative B staging: thread t async-copies 32B of column (t>>1).
  const int scol  = tid >> 1;               // 0..127
  const int spart = (tid & 1) * 16;         // 0 or 16
  const _Float16* bsrc = BT + (size_t)(st * 128 + scol) * (size_t)ldb + spart;
  // Low 32 bits of the generic address of an LDS object are the LDS byte
  // offset (ISA 00_overview §10.2 aperture mapping).
  const unsigned sdst =
      (unsigned)(size_t)&bs[0] + (unsigned)((scol * 32 + spart) * 2);

  // Prologue: stage slab 0.
  async_copy32(sdst, bsrc);
  wait_async0();
  __syncthreads();

  v8f acc[8] = {};
  for (int k0 = 0; k0 < K; k0 += 32) {
    const int cur = (k0 >> 5) & 1;
    const bool more = (k0 + 32) < K;

    // Issue next slab's async load; it lands in LDS under the WMMAs.
    if (more) {
      async_copy32(sdst + (unsigned)((cur ^ 1) * SLAB * 2), bsrc + k0 + 32);
      __builtin_prefetch(A + arow + k0 + 64, 0, 3);
    }

    // A fragment (contiguous K order to match B: lanes 0-15 K 0-15,
    // lanes 16-31 K 16-31).
    v8h alo = *(const v8h*)(A + arow + k0 + kh * 16);
    v8h ahi = *(const v8h*)(A + arow + k0 + 8 + kh * 16);
    v16h af = cat16(alo, ahi);

    // Software-pipelined B fragments from LDS (lane = column; lanes 0-15 hold
    // K 0-15, lanes 16-31 K 16-31).
    const _Float16* bbase = bs + cur * SLAB + l15 * 32 + kh * 16;
    v16h bf[2];
    {
      v8h lo = *(const v8h*)bbase;
      v8h hi = *(const v8h*)(bbase + 8);
      bf[0] = cat16(lo, hi);
    }
#pragma unroll
    for (int t = 0; t < 8; ++t) {
      if (t < 7) {
        const _Float16* p = bbase + (t + 1) * 16 * 32;
        v8h lo = *(const v8h*)p;
        v8h hi = *(const v8h*)(p + 8);
        bf[(t + 1) & 1] = cat16(lo, hi);
      }
      acc[t] = wmma_f16(af, bf[t & 1], acc[t]);
    }

    // Own async issues complete, then block-wide visibility.
    wait_async0();
    __syncthreads();
  }

#pragma unroll
  for (int t = 0; t < 8; ++t) {
    const int ncol = st * 128 + t * 16 + l15;
    const float bv = bias ? bias[ncol] : 0.0f;
#pragma unroll
    for (int r = 0; r < 8; ++r) {
      const int m = mt * 16 + kh * 8 + r;   // C layout: VGPR r -> row r + 8*(lane>>4)
      float v = acc[t][r] * scale + bv;
      if (resid) v += resid[(size_t)m * ldc + ncol];
      if (do_gelu) v = gelu_exact(v);
      if (C)   C[(size_t)m * ldc + ncol] = v;
      if (C16) C16[(size_t)m * ldc + ncol] = (_Float16)v;
    }
  }
}

// ---------------------------------------------------------------------------
// Batched scores: S[bh][48][2048] = scale * QR[bh][48][32] @ K16[b,:,h*32:+32]^T
// Single K-step of 32; blockIdx.y = bh.
// ---------------------------------------------------------------------------
__global__ __launch_bounds__(256) void scores_wmma_kernel(
    const _Float16* __restrict__ QR, const _Float16* __restrict__ K16,
    float* __restrict__ S, float scale)
{
  const int bh = blockIdx.y;
  const int b = bh >> 3, h = bh & 7;
  const int wave = (blockIdx.x * blockDim.x + threadIdx.x) >> 5;
  const int lane = threadIdx.x & 31;
  if (wave >= 3 * 32) return;                // 3 M-tiles x 32 strips
  const int mt = wave / 32, st = wave % 32;
  const int l15 = lane & 15, kh = lane >> 4;

  const _Float16* A  = QR + (size_t)bh * CUP * CDK;
  const _Float16* BT = K16 + (size_t)b * CL * CD + h * CDK;

  const int arow = mt * 16 + l15;
  v8h alo = *(const v8h*)(A + arow * CDK + kh * 16);
  v8h ahi = *(const v8h*)(A + arow * CDK + 8 + kh * 16);
  v16h af = cat16(alo, ahi);

  v8f acc[4] = {};
#pragma unroll
  for (int t = 0; t < 4; ++t) {
    const int ncol = (st * 4 + t) * 16 + l15;          // key index
    const _Float16* bp = BT + (size_t)ncol * CD + kh * 16;
    v8h blo = *(const v8h*)bp;
    v8h bhi = *(const v8h*)(bp + 8);
    acc[t] = wmma_f16(af, cat16(blo, bhi), acc[t]);
  }
#pragma unroll
  for (int t = 0; t < 4; ++t) {
    const int ncol = (st * 4 + t) * 16 + l15;
#pragma unroll
    for (int r = 0; r < 8; ++r) {
      const int m = mt * 16 + kh * 8 + r;
      S[((size_t)bh * CUP + m) * CL + ncol] = acc[t][r] * scale;
    }
  }
}

// ---------------------------------------------------------------------------
// Elementwise / support kernels
// ---------------------------------------------------------------------------
__global__ void f32tof16_kernel(const float* __restrict__ in,
                                _Float16* __restrict__ out, int n) {
  int i = blockIdx.x * blockDim.x + threadIdx.x;
  if (i < n) out[i] = (_Float16)in[i];
}

__global__ void wtrans_kernel(const float* __restrict__ W,
                              _Float16* __restrict__ WT, int K, int N) {
  int i = blockIdx.x * blockDim.x + threadIdx.x;
  if (i >= K * N) return;
  int n = i / K, k = i - n * K;
  WT[i] = (_Float16)W[(size_t)k * N + n];   // WT[n][k] = W[k][n]
}

// Embedding: conv1d(k=3, circular pad) + sinusoidal pos emb + mark @ temp_w
__global__ void embed_kernel(const float* __restrict__ x,
                             const float* __restrict__ mark,
                             const float* __restrict__ convw,
                             const float* __restrict__ tempw,
                             float* __restrict__ out) {
  int i = blockIdx.x * blockDim.x + threadIdx.x;
  if (i >= CTOK * CD) return;
  int d  = i & (CD - 1);
  int bl = i >> 8;
  int l  = bl & (CL - 1);
  int b  = bl >> 11;
  float acc = 0.0f;
#pragma unroll
  for (int w = 0; w < 3; ++w) {
    int src = l + w;                        // index into padded sequence
    int ls = (src == 0) ? (CL - 1) : ((src == CL + 1) ? 0 : src - 1);
    const float* xr = x + ((size_t)(b * CL + ls)) * CCO;
    const float* cw = convw + (size_t)(w * CCO) * CD + d;
#pragma unroll
    for (int c = 0; c < CCO; ++c) acc += xr[c] * cw[(size_t)c * CD];
  }
  float div = expf(-(float)(d & ~1) * (logf(10000.0f) / (float)CD));
  float ang = (float)l * div;
  acc += (d & 1) ? cosf(ang) : sinf(ang);
  const float* mr = mark + (size_t)bl * 4;
#pragma unroll
  for (int m = 0; m < 4; ++m) acc += mr[m] * tempw[m * CD + d];
  out[i] = acc;
}

// M-measure: one wave per (b,h,l); U=40 sampled keys via hash.
__global__ __launch_bounds__(256) void prob_m_kernel(
    const float* __restrict__ Q, const float* __restrict__ K,
    float* __restrict__ Mout, unsigned seed) {
  int wave = (blockIdx.x * blockDim.x + threadIdx.x) >> 5;
  int lane = threadIdx.x & 31;
  if (wave >= CBH * CL) return;
  int l = wave & (CL - 1);
  int bh = wave >> 11;
  int b = bh >> 3, h = bh & 7;
  const float* q = Q + ((size_t)(b * CL + l)) * CD + h * CDK;
  float qr[CDK];
#pragma unroll
  for (int d = 0; d < CDK; ++d) qr[d] = q[d];
  float mx = -FLT_MAX, sm = 0.0f;
  for (int u = lane; u < CUU; u += 32) {
    unsigned r = mix_hash(seed * 0x9e3779b9u ^ (unsigned)l * 0x85ebca77u ^
                          (unsigned)u * 0xc2b2ae3du);
    int kl = (int)(r & (CL - 1));
    const float* kp = K + ((size_t)(b * CL + kl)) * CD + h * CDK;
    float s = 0.0f;
#pragma unroll
    for (int d = 0; d < CDK; ++d) s += qr[d] * kp[d];
    mx = fmaxf(mx, s);
    sm += s;
  }
#pragma unroll
  for (int o = 16; o > 0; o >>= 1) {
    mx = fmaxf(mx, __shfl_xor(mx, o, 32));
    sm += __shfl_xor(sm, o, 32);
  }
  if (lane == 0) Mout[wave] = mx - sm * (1.0f / (float)CL);
}

// Iterative top-40 per (b,h) using LDS.
__global__ __launch_bounds__(256) void topk_kernel(const float* __restrict__ M,
                                                   int* __restrict__ topi) {
  __shared__ float sv[CL];
  __shared__ float smax[256];
  __shared__ int   sidx[256];
  const int bh = blockIdx.x, t = threadIdx.x;
  for (int i = t; i < CL; i += 256) sv[i] = M[(size_t)bh * CL + i];
  __syncthreads();
  for (int it = 0; it < CUU; ++it) {
    float best = -FLT_MAX; int bi = 0;
    for (int i = t; i < CL; i += 256) {
      float v = sv[i];
      if (v > best) { best = v; bi = i; }
    }
    smax[t] = best; sidx[t] = bi;
    __syncthreads();
    for (int s = 128; s > 0; s >>= 1) {
      if (t < s) {
        if (smax[t + s] > smax[t] ||
            (smax[t + s] == smax[t] && sidx[t + s] < sidx[t])) {
          smax[t] = smax[t + s]; sidx[t] = sidx[t + s];
        }
      }
      __syncthreads();
    }
    if (t == 0) { topi[bh * CUU + it] = sidx[0]; sv[sidx[0]] = -FLT_MAX; }
    __syncthreads();
  }
}

__global__ void gather_q_kernel(const float* __restrict__ Q,
                                const int* __restrict__ topi,
                                _Float16* __restrict__ qr) {
  int i = blockIdx.x * blockDim.x + threadIdx.x;
  if (i >= CBH * CUP * CDK) return;
  int d = i & 31;
  int u = (i >> 5) % CUP;
  int bh = i / (CUP * CDK);
  float v = 0.0f;
  if (u < CUU) {
    int b = bh >> 3, h = bh & 7;
    int l = topi[bh * CUU + u];
    v = Q[((size_t)(b * CL + l)) * CD + h * CDK + d];
  }
  qr[i] = (_Float16)v;
}

// Row softmax with optional causal mask (mask keys beyond the query position).
__global__ __launch_bounds__(256) void softmax_kernel(
    float* __restrict__ S, const int* __restrict__ topi, int causal) {
  int wave = (blockIdx.x * blockDim.x + threadIdx.x) >> 5;
  int lane = threadIdx.x & 31;
  if (wave >= CBH * CUU) return;
  int u = wave % CUU, bh = wave / CUU;
  float* row = S + ((size_t)bh * CUP + u) * CL;
  int lim = causal ? topi[bh * CUU + u] : (CL - 1);
  float mx = -FLT_MAX;
  for (int i = lane; i < CL; i += 32) {
    float v = (i <= lim) ? row[i] : -FLT_MAX;
    mx = fmaxf(mx, v);
  }
#pragma unroll
  for (int o = 16; o > 0; o >>= 1) mx = fmaxf(mx, __shfl_xor(mx, o, 32));
  float sm = 0.0f;
  for (int i = lane; i < CL; i += 32) {
    float e = (i <= lim) ? expf(row[i] - mx) : 0.0f;
    row[i] = e;
    sm += e;
  }
#pragma unroll
  for (int o = 16; o > 0; o >>= 1) sm += __shfl_xor(sm, o, 32);
  float inv = 1.0f / sm;
  for (int i = lane; i < CL; i += 32) row[i] *= inv;
}

__global__ void meanv_kernel(const float* __restrict__ V, float* __restrict__ mv) {
  int i = blockIdx.x * blockDim.x + threadIdx.x;
  if (i >= CBH * CDK) return;
  int d = i & 31, bh = i >> 5, b = bh >> 3, h = bh & 7;
  const float* vp = V + (size_t)b * CL * CD + h * CDK + d;
  float s = 0.0f;
  for (int l = 0; l < CL; ++l) s += vp[(size_t)l * CD];
  mv[i] = s * (1.0f / (float)CL);
}

__global__ void ctx_mean_kernel(const float* __restrict__ mv,
                                float* __restrict__ ctx) {
  int i = blockIdx.x * blockDim.x + threadIdx.x;
  if (i >= CTOK * CD) return;
  int c = i & (CD - 1);
  int bl = i >> 8;
  int b = bl >> 11;
  int h = c >> 5, d = c & 31;
  ctx[i] = mv[((b << 3) + h) * CDK + d];
}

__global__ void ctx_cumsum_kernel(const float* __restrict__ V,
                                  float* __restrict__ ctx) {
  int i = blockIdx.x * blockDim.x + threadIdx.x;
  if (i >= CBH * CDK) return;
  int d = i & 31, bh = i >> 5, b = bh >> 3, h = bh & 7;
  size_t base = (size_t)b * CL * CD + h * CDK + d;
  float s = 0.0f;
  for (int l = 0; l < CL; ++l) {
    s += V[base + (size_t)l * CD];
    ctx[base + (size_t)l * CD] = s;
  }
}

// upd = attn @ V, then scatter to ctx at the top-k rows. One wave per (bh,u).
__global__ __launch_bounds__(256) void upd_scatter_kernel(
    const float* __restrict__ S, const float* __restrict__ V,
    const int* __restrict__ topi, float* __restrict__ ctx) {
  int wave = (blockIdx.x * blockDim.x + threadIdx.x) >> 5;
  int lane = threadIdx.x & 31;
  if (wave >= CBH * CUU) return;
  int u = wave % CUU, bh = wave / CUU, b = bh >> 3, h = bh & 7;
  const float* row = S + ((size_t)bh * CUP + u) * CL;
  const float* vp = V + (size_t)b * CL * CD + h * CDK + lane;
  float acc = 0.0f;
  for (int lk = 0; lk < CL; ++lk) acc += row[lk] * vp[(size_t)lk * CD];
  int l = topi[bh * CUU + u];
  ctx[((size_t)(b * CL + l)) * CD + h * CDK + lane] = acc;
}

// LayerNorm over d_model=256 (one wave per row), optional residual add first.
__global__ __launch_bounds__(256) void layernorm_kernel(
    const float* __restrict__ X, const float* __restrict__ R,
    const float* __restrict__ g, const float* __restrict__ beta,
    float* __restrict__ out) {
  int wave = (blockIdx.x * blockDim.x + threadIdx.x) >> 5;
  int lane = threadIdx.x & 31;
  if (wave >= CTOK) return;
  const float* xr = X + (size_t)wave * CD;
  const float* rr = R ? R + (size_t)wave * CD : nullptr;
  float x[8];
  float s = 0.0f, s2 = 0.0f;
#pragma unroll
  for (int i = 0; i < 8; ++i) {
    int d = lane + 32 * i;
    float v = xr[d];
    if (rr) v += rr[d];
    x[i] = v; s += v; s2 += v * v;
  }
#pragma unroll
  for (int o = 16; o > 0; o >>= 1) {
    s  += __shfl_xor(s, o, 32);
    s2 += __shfl_xor(s2, o, 32);
  }
  float mean = s * (1.0f / CD);
  float var  = s2 * (1.0f / CD) - mean * mean;
  float rstd = rsqrtf(var + 1e-5f);
  float* orow = out + (size_t)wave * CD;
#pragma unroll
  for (int i = 0; i < 8; ++i) {
    int d = lane + 32 * i;
    orow[d] = (x[i] - mean) * rstd * g[d] + beta[d];
  }
}

// Final projection, last PRED_LEN positions only.
__global__ void proj_kernel(const float* __restrict__ X,
                            const float* __restrict__ W,
                            const float* __restrict__ bias,
                            float* __restrict__ out) {
  int i = blockIdx.x * blockDim.x + threadIdx.x;
  if (i >= CB * CPRED * CCO) return;
  int o = i % CCO;
  int t = i / CCO;
  int lp = t % CPRED, b = t / CPRED;
  const float* row = X + ((size_t)(b * CL + (CL - CPRED) + lp)) * CD;
  float acc = bias[o];
  for (int d = 0; d < CD; ++d) acc += row[d] * W[d * CCO + o];
  out[i] = acc;
}

// ---------------------------------------------------------------------------
// Host orchestration
// ---------------------------------------------------------------------------
static inline int divup(int a, int b) { return (a + b - 1) / b; }

static inline void launch_gemm(hipStream_t s, const _Float16* A, int lda,
                               const _Float16* BT, int ldb, const float* bias,
                               const float* resid, float* C, _Float16* C16,
                               int ldc, int M, int N, int K, float scale,
                               int gelu) {
  int blocks = (M >> 7) * (N >> 7);   // one block per 128x128 tile
  gemm_wmma_kernel<<<blocks, 256, 0, s>>>(A, lda, BT, ldb, bias, resid, C, C16,
                                          ldc, M, N, K, scale, gelu);
}

struct Scratch {
  float *ENC, *DEC, *TMP, *Q, *Kb, *V, *CTX, *SCORES, *MBUF, *MEANV;
  int* TOPI;
  _Float16 *A16, *B16, *K16, *QR16;
};

static void run_attention(hipStream_t s, const Scratch& w, const float* Xq,
                          const float* Xkv, const _Float16* wq,
                          const _Float16* wk, const _Float16* wv,
                          const _Float16* wo, const float* bq, const float* bk,
                          const float* bv, const float* bo, int causal,
                          unsigned seed, const float* resid, float* out) {
  const int NACT = CTOK * CD;
  f32tof16_kernel<<<divup(NACT, 256), 256, 0, s>>>(Xq, w.A16, NACT);
  launch_gemm(s, w.A16, CD, wq, CD, bq, nullptr, w.Q, nullptr, CD, CTOK, CD,
              CD, 1.f, 0);
  if (Xkv != Xq)
    f32tof16_kernel<<<divup(NACT, 256), 256, 0, s>>>(Xkv, w.A16, NACT);
  // K projection also emits the f16 copy used by the scores WMMA kernel.
  launch_gemm(s, w.A16, CD, wk, CD, bk, nullptr, w.Kb, w.K16, CD, CTOK, CD,
              CD, 1.f, 0);
  launch_gemm(s, w.A16, CD, wv, CD, bv, nullptr, w.V, nullptr, CD, CTOK, CD,
              CD, 1.f, 0);

  prob_m_kernel<<<divup(CBH * CL * 32, 256), 256, 0, s>>>(w.Q, w.Kb, w.MBUF, seed);
  topk_kernel<<<CBH, 256, 0, s>>>(w.MBUF, w.TOPI);
  gather_q_kernel<<<divup(CBH * CUP * CDK, 256), 256, 0, s>>>(w.Q, w.TOPI, w.QR16);
  scores_wmma_kernel<<<dim3(12, CBH), 256, 0, s>>>(w.QR16, w.K16, w.SCORES,
                                                   0.17677669529663687f);
  softmax_kernel<<<divup(CBH * CUU * 32, 256), 256, 0, s>>>(w.SCORES, w.TOPI, causal);
  if (causal) {
    ctx_cumsum_kernel<<<divup(CBH * CDK, 256), 256, 0, s>>>(w.V, w.CTX);
  } else {
    meanv_kernel<<<divup(CBH * CDK, 256), 256, 0, s>>>(w.V, w.MEANV);
    ctx_mean_kernel<<<divup(NACT, 256), 256, 0, s>>>(w.MEANV, w.CTX);
  }
  upd_scatter_kernel<<<divup(CBH * CUU * 32, 256), 256, 0, s>>>(w.SCORES, w.V,
                                                               w.TOPI, w.CTX);
  f32tof16_kernel<<<divup(NACT, 256), 256, 0, s>>>(w.CTX, w.A16, NACT);
  launch_gemm(s, w.A16, CD, wo, CD, bo, resid, out, nullptr, CD, CTOK, CD, CD,
              1.f, 0);
}

static void run_ffn(hipStream_t s, const Scratch& w, const float* X,
                    const _Float16* w1t, const float* b1, const _Float16* w2t,
                    const float* b2, float* out) {
  f32tof16_kernel<<<divup(CTOK * CD, 256), 256, 0, s>>>(X, w.A16, CTOK * CD);
  // FFN1 emits only the f16 intermediate (no f32 store at all).
  launch_gemm(s, w.A16, CD, w1t, CD, b1, nullptr, nullptr, w.B16, CDF, CTOK,
              CDF, CD, 1.f, 1);
  launch_gemm(s, w.B16, CDF, w2t, CDF, b2, X, out, nullptr, CD, CTOK, CD, CDF,
              1.f, 0);
}

// d_in index map (sorted-key jax tree flatten). See file header for details.
enum {
  WT_E0_WQ = 0, WT_E0_WK, WT_E0_WV, WT_E0_WO, WT_E0_W1, WT_E0_W2,
  WT_E1_WQ, WT_E1_WK, WT_E1_WV, WT_E1_WO, WT_E1_W1, WT_E1_W2,
  WT_DS_WQ, WT_DS_WK, WT_DS_WV, WT_DS_WO,
  WT_DC_WQ, WT_DC_WK, WT_DC_WV, WT_DC_WO,
  WT_D_W1, WT_D_W2,
  WT_COUNT
};

extern "C" void kernel_launch(void* const* d_in, const int* in_sizes, int n_in,
                              void* d_out, int out_size, void* d_ws,
                              size_t ws_size, hipStream_t stream) {
  (void)in_sizes; (void)out_size; (void)ws_size;
  const float* P[80];
  for (int i = 0; i < n_in && i < 80; ++i) P[i] = (const float*)d_in[i];

  // Workspace carve-out (deterministic every call).
  char* ws = (char*)d_ws;
  size_t off = 0;
  auto take = [&](size_t bytes) -> void* {
    void* p = ws + off;
    off += (bytes + 255) & ~(size_t)255;
    return p;
  };
  Scratch w;
  w.ENC   = (float*)take((size_t)CTOK * CD * 4);
  w.DEC   = (float*)take((size_t)CTOK * CD * 4);
  w.TMP   = (float*)take((size_t)CTOK * CD * 4);
  w.Q     = (float*)take((size_t)CTOK * CD * 4);
  w.Kb    = (float*)take((size_t)CTOK * CD * 4);
  w.V     = (float*)take((size_t)CTOK * CD * 4);
  w.CTX   = (float*)take((size_t)CTOK * CD * 4);
  w.SCORES= (float*)take((size_t)CBH * CUP * CL * 4);
  w.MBUF  = (float*)take((size_t)CBH * CL * 4);
  w.MEANV = (float*)take((size_t)CBH * CDK * 4);
  w.TOPI  = (int*)take((size_t)CBH * CUU * 4);
  w.A16   = (_Float16*)take((size_t)CTOK * CD * 2);
  w.B16   = (_Float16*)take((size_t)CTOK * CDF * 2);
  w.K16   = (_Float16*)take((size_t)CTOK * CD * 2);
  w.QR16  = (_Float16*)take((size_t)CBH * CUP * CDK * 2);

  // Weight transpose+convert (f32 [K][N] -> f16 [N][K]).
  static const struct { int idx, K, N; } wspec[WT_COUNT] = {
      {38, 256, 256}, {36, 256, 256}, {39, 256, 256}, {37, 256, 256},
      {46, 256, 1024}, {47, 1024, 256},
      {54, 256, 256}, {52, 256, 256}, {55, 256, 256}, {53, 256, 256},
      {62, 256, 1024}, {63, 1024, 256},
      {24, 256, 256}, {22, 256, 256}, {25, 256, 256}, {23, 256, 256},
      {10, 256, 256}, {8, 256, 256}, {11, 256, 256}, {9, 256, 256},
      {26, 256, 1024}, {27, 1024, 256},
  };
  _Float16* wt[WT_COUNT];
  for (int i = 0; i < WT_COUNT; ++i) {
    int n = wspec[i].K * wspec[i].N;
    wt[i] = (_Float16*)take((size_t)n * 2);
    wtrans_kernel<<<divup(n, 256), 256, 0, stream>>>(P[wspec[i].idx], wt[i],
                                                     wspec[i].K, wspec[i].N);
  }

  const int NACT = CTOK * CD;
  const int LN_BLOCKS = divup(CTOK * 32, 256);

  // ===== Encoder =====
  embed_kernel<<<divup(NACT, 256), 256, 0, stream>>>(P[69], P[71], P[30], P[31], w.ENC);
  for (int i = 0; i < 2; ++i) {
    const int base = 32 + i * 16;  // bk,bo,bq,bv,wk,wo,wq,wv,b1,b2,ln1_b,ln1_g,ln2_b,ln2_g,w1,w2
    run_attention(stream, w, w.ENC, w.ENC, wt[WT_E0_WQ + i * 6],
                  wt[WT_E0_WK + i * 6], wt[WT_E0_WV + i * 6],
                  wt[WT_E0_WO + i * 6], P[base + 2], P[base + 0], P[base + 3],
                  P[base + 1], 0, (unsigned)i, w.ENC, w.TMP);
    layernorm_kernel<<<LN_BLOCKS, 256, 0, stream>>>(w.TMP, nullptr, P[base + 11],
                                                    P[base + 10], w.TMP);
    run_ffn(stream, w, w.TMP, wt[WT_E0_W1 + i * 6], P[base + 8],
            wt[WT_E0_W2 + i * 6], P[base + 9], w.Q);
    layernorm_kernel<<<LN_BLOCKS, 256, 0, stream>>>(w.Q, nullptr, P[base + 13],
                                                    P[base + 12], w.ENC);
  }
  layernorm_kernel<<<LN_BLOCKS, 256, 0, stream>>>(w.ENC, nullptr, P[65], P[64], w.ENC);

  // ===== Decoder =====
  embed_kernel<<<divup(NACT, 256), 256, 0, stream>>>(P[68], P[70], P[0], P[1], w.DEC);
  // self-attention (causal)
  run_attention(stream, w, w.DEC, w.DEC, wt[WT_DS_WQ], wt[WT_DS_WK],
                wt[WT_DS_WV], wt[WT_DS_WO], P[20], P[18], P[21], P[19], 1, 100u,
                w.DEC, w.TMP);
  layernorm_kernel<<<LN_BLOCKS, 256, 0, stream>>>(w.TMP, nullptr, P[13], P[12], w.TMP);
  // cross-attention (queries from TMP, keys/values from encoder output)
  run_attention(stream, w, w.TMP, w.ENC, wt[WT_DC_WQ], wt[WT_DC_WK],
                wt[WT_DC_WV], wt[WT_DC_WO], P[6], P[4], P[7], P[5], 0, 101u,
                w.TMP, w.TMP);
  layernorm_kernel<<<LN_BLOCKS, 256, 0, stream>>>(w.TMP, nullptr, P[15], P[14], w.TMP);
  run_ffn(stream, w, w.TMP, wt[WT_D_W1], P[2], wt[WT_D_W2], P[3], w.Q);
  layernorm_kernel<<<LN_BLOCKS, 256, 0, stream>>>(w.Q, nullptr, P[17], P[16], w.DEC);
  layernorm_kernel<<<LN_BLOCKS, 256, 0, stream>>>(w.DEC, nullptr, P[29], P[28], w.DEC);

  proj_kernel<<<divup(CB * CPRED * CCO, 256), 256, 0, stream>>>(
      w.DEC, P[67], P[66], (float*)d_out);
}
